// StructureEncoder_60498909331494
// MI455X (gfx1250) — matrix operand order repaired and checked by reading, not compiled, not run on hardware
//
#include <hip/hip_runtime.h>
#include <hip/hip_bf16.h>

#define NATOMS 100000
#define MNBR   12
#define AFL    64
#define NROWS  (NATOMS * MNBR)     // 1,200,000
#define INFULL 169
#define KEMB   92
#define NCRY   2000
#define APC    50
#define ODIM   128
#define EPSBN  1e-5f

// Fragment-order weight layouts (dwords):
//   conv : [6 ksteps][8 col tiles][32 lanes][8 dwords]  = 12288 dwords (48 KB)
//   embed: [3 ksteps][4 col tiles][32 lanes][8 dwords]  =  3072 dwords (12 KB)
#define WF_DW (6 * 8 * 32 * 8)
#define WE_DW (3 * 4 * 32 * 8)

typedef __attribute__((ext_vector_type(16))) __bf16 v16bf;
typedef __attribute__((ext_vector_type(8)))  float  v8f;

union Frag16 { v16bf v; unsigned u[8]; uint4 q[2]; };

// Pack two f32 into one bf16x2 dword with a single v_perm_b32 (truncation):
// low half = a, high half = b.
__device__ __forceinline__ unsigned pack_bf16(float a, float b) {
  return __builtin_amdgcn_perm(__float_as_uint(b), __float_as_uint(a), 0x07060302u);
}

__device__ __forceinline__ float softplusf(float v) {
  return v > 20.f ? v : log1pf(__expf(v));
}

// ---------- A-fragment builders (16-bit A layout: lane=row, K halves by lane half) ----------

__device__ __forceinline__ Frag16 frag_vec(const float* __restrict__ p, int koff) {
  Frag16 a;
  float4 A = *(const float4*)(p + koff);
  float4 B = *(const float4*)(p + koff + 4);
  float4 C = *(const float4*)(p + koff + 16);
  float4 D = *(const float4*)(p + koff + 20);
  a.u[0] = pack_bf16(A.x, A.y); a.u[1] = pack_bf16(A.z, A.w);
  a.u[2] = pack_bf16(B.x, B.y); a.u[3] = pack_bf16(B.z, B.w);
  a.u[4] = pack_bf16(C.x, C.y); a.u[5] = pack_bf16(C.z, C.w);
  a.u[6] = pack_bf16(D.x, D.y); a.u[7] = pack_bf16(D.z, D.w);
  return a;
}

__device__ __forceinline__ Frag16 frag_scalar(const float* __restrict__ p, int koff) {
  Frag16 a;
#pragma unroll
  for (int j = 0; j < 4; ++j) {
    a.u[j]     = pack_bf16(p[koff + 2 * j],      p[koff + 2 * j + 1]);
    a.u[4 + j] = pack_bf16(p[koff + 16 + 2 * j], p[koff + 17 + 2 * j]);
  }
  return a;
}

__device__ __forceinline__ float ldz(const float* __restrict__ p, int i, int n) {
  int j = i < n ? i : n - 1;
  float v = p[j];
  return i < n ? v : 0.f;
}

__device__ __forceinline__ Frag16 frag_clamped(const float* __restrict__ p, int koff, int n) {
  Frag16 a;
#pragma unroll
  for (int j = 0; j < 4; ++j) {
    a.u[j]     = pack_bf16(ldz(p, koff + 2 * j, n),      ldz(p, koff + 2 * j + 1, n));
    a.u[4 + j] = pack_bf16(ldz(p, koff + 16 + 2 * j, n), ldz(p, koff + 17 + 2 * j, n));
  }
  return a;
}

// B fragments from LDS (fragment-order layout) + WMMA across NT column tiles.
// wbase = per-lane base (lane*8 dwords); fragment (ks, t) at +(ks*NT+t)*256 dwords.
template <int NT>
__device__ __forceinline__ void wmma_row(v8f* acc, const Frag16& a,
                                         const unsigned* wbase, int ks) {
#pragma unroll
  for (int t = 0; t < NT; ++t) {
    Frag16 b;
    const unsigned* p = wbase + (ks * NT + t) * 256;
    b.q[0] = *(const uint4*)(p);
    b.q[1] = *(const uint4*)(p + 4);
    acc[t] = __builtin_amdgcn_wmma_f32_16x16x32_bf16(false, a.v, false, b.v,
                                                     (short)0, acc[t], false, false);
  }
}

// ---------------- weight conversion: write fragment-order bf16 layouts ----------------

__global__ void k_cvt_wemb(const float* __restrict__ We, unsigned* __restrict__ WeT) {
  for (int i = blockIdx.x * blockDim.x + threadIdx.x; i < WE_DW; i += gridDim.x * blockDim.x) {
    int j    = i & 7;
    int lane = (i >> 3) & 31;
    int t    = (i >> 8) & 3;
    int ks   = i >> 10;                      // 0..2
    int col  = t * 16 + (lane & 15);
    int k    = ks * 32 + (lane >> 4) * 8 + (j >> 2) * 16 + 2 * (j & 3);
    float v0 = (k     < KEMB) ? We[k * AFL + col]       : 0.f;
    float v1 = (k + 1 < KEMB) ? We[(k + 1) * AFL + col] : 0.f;
    WeT[i] = pack_bf16(v0, v1);
  }
}

__global__ void k_cvt_wfull(const float* __restrict__ Wf, unsigned* __restrict__ WfT) {
  for (int i = blockIdx.x * blockDim.x + threadIdx.x; i < WF_DW; i += gridDim.x * blockDim.x) {
    int j    = i & 7;
    int lane = (i >> 3) & 31;
    int t    = (i >> 8) & 7;
    int ks   = i >> 11;                      // 0..5
    int col  = t * 16 + (lane & 15);
    int k    = ks * 32 + (lane >> 4) * 8 + (j >> 2) * 16 + 2 * (j & 3);
    float v0 = (k     < INFULL) ? Wf[k * (2 * AFL) + col]       : 0.f;
    float v1 = (k + 1 < INFULL) ? Wf[(k + 1) * (2 * AFL) + col] : 0.f;
    WfT[i] = pack_bf16(v0, v1);
  }
}

// ---------------- embedding GEMM: x = atom_fea @ W_embed + b ----------------

__global__ void k_embed(const float* __restrict__ af, const unsigned* __restrict__ WeT,
                        const float* __restrict__ be, float* __restrict__ x) {
  __shared__ __align__(16) unsigned sW[WE_DW];   // 12 KB
  int tid = threadIdx.x;
  {
    const uint4* s = (const uint4*)WeT;
    uint4* d = (uint4*)sW;
    for (int i = tid; i < WE_DW / 4; i += 256) d[i] = s[i];
  }
  __syncthreads();

  int lane = tid & 31, wave = tid >> 5;
  int ln = lane & 15, half = lane >> 4, koff = half * 8;
  int row0 = blockIdx.x * 128 + wave * 16;
  int garow = row0 + ln;
  int gclamp = garow < NATOMS ? garow : NATOMS - 1;   // OOB rows compute garbage, never stored
  const float* arow = af + (size_t)gclamp * KEMB;     // 92 floats = 368B -> 16B aligned
  const unsigned* wbase = sW + lane * 8;

  v8f acc[4] = {};
  wmma_row<4>(acc, frag_vec(arow, koff),              wbase, 0);
  wmma_row<4>(acc, frag_vec(arow + 32, koff),         wbase, 1);
  wmma_row<4>(acc, frag_clamped(arow + 64, koff, 28), wbase, 2);  // k 64..91 valid

  int rb = row0 + half * 8;
#pragma unroll
  for (int t = 0; t < 4; ++t) {
    int col = t * 16 + ln;
    float bb = be[col];
#pragma unroll
    for (int r = 0; r < 8; ++r) {
      int row = rb + r;
      if (row < NATOMS) x[(size_t)row * AFL + col] = acc[t][r] + bb;
    }
  }
}

// ---------------- conv GEMM (two passes: stats / apply) ----------------

template <bool APPLY>
__global__ void k_conv(const float* __restrict__ x, const int* __restrict__ idx,
                       const float* __restrict__ nf, const unsigned* __restrict__ WfT,
                       const float* __restrict__ bfull,
                       const float* __restrict__ A1, const float* __restrict__ B1,
                       float* __restrict__ gS, float* __restrict__ gQ,
                       float* __restrict__ ns) {
  __shared__ __align__(16) unsigned sW[WF_DW];   // 48 KB
  __shared__ float sS[2 * AFL], sQ[2 * AFL];
  int tid = threadIdx.x;
  if (!APPLY && tid < 2 * AFL) { sS[tid] = 0.f; sQ[tid] = 0.f; }
  {
    const uint4* s = (const uint4*)WfT;
    uint4* d = (uint4*)sW;
    for (int i = tid; i < WF_DW / 4; i += 256) d[i] = s[i];
  }
  __syncthreads();

  int lane = tid & 31, wave = tid >> 5;
  int ln = lane & 15, half = lane >> 4, koff = half * 8;
  int row0 = blockIdx.x * 128 + wave * 16;
  int g = row0 + ln;                          // flattened (atom, neighbor) row for A
  unsigned n = (unsigned)g / MNBR;            // const-div -> mul/shift
  const float* xrow  = x + (size_t)n * AFL;                 // 256B rows, aligned
  const float* xnrow = x + (size_t)idx[g] * AFL;
  const float* nrow  = nf + (size_t)g * 41;                 // unaligned
  const unsigned* wbase = sW + lane * 8;

  v8f acc[8] = {};
  wmma_row<8>(acc, frag_vec(xrow, koff),              wbase, 0);   // k   0.. 63: self
  wmma_row<8>(acc, frag_vec(xrow + 32, koff),         wbase, 1);
  wmma_row<8>(acc, frag_vec(xnrow, koff),             wbase, 2);   // k  64..127: neighbor
  wmma_row<8>(acc, frag_vec(xnrow + 32, koff),        wbase, 3);
  wmma_row<8>(acc, frag_scalar(nrow, koff),           wbase, 4);   // k 128..159: bond fea
  wmma_row<8>(acc, frag_clamped(nrow + 32, koff, 9),  wbase, 5);   // k 160..168 valid

  int rb = row0 + half * 8;                   // first output row held by this lane
  if (!APPLY) {
    // per-column sum / sumsq of gated = acc + bias; wave -> LDS -> global
#pragma unroll
    for (int t = 0; t < 8; ++t) {
      int col = t * 16 + ln;
      float bb = bfull[col];
      float s = 0.f, q = 0.f;
#pragma unroll
      for (int r = 0; r < 8; ++r) { float v = acc[t][r] + bb; s += v; q += v * v; }
      s += __shfl_xor(s, 16, 32);
      q += __shfl_xor(q, 16, 32);
      if (half == 0) { atomicAdd(&sS[col], s); atomicAdd(&sQ[col], q); }
    }
    __syncthreads();
    if (tid < 2 * AFL) { atomicAdd(&gS[tid], sS[tid]); atomicAdd(&gQ[tid], sQ[tid]); }
  } else {
    // BN1 affine (bias folded into B1) + sigmoid*softplus, reduce over neighbors
    unsigned at0 = (unsigned)rb / MNBR;
    int rm0 = rb - (int)at0 * MNBR;
#pragma unroll
    for (int t = 0; t < 4; ++t) {
      int colF = t * 16 + ln;
      float aF = A1[colF],       bF = B1[colF];
      float aC = A1[colF + AFL], bC = B1[colF + AFL];
      unsigned atom = at0; int rm = rm0; float accm = 0.f;
#pragma unroll
      for (int r = 0; r < 8; ++r) {
        float filt = 1.f / (1.f + __expf(-(aF * acc[t][r] + bF)));
        float core = softplusf(aC * acc[t + 4][r] + bC);
        accm += filt * core;
        if (++rm == MNBR) {                    // finished an atom's 12 neighbors
          atomicAdd(&ns[atom * AFL + colF], accm);
          accm = 0.f; rm = 0; ++atom;
        }
      }
      if (rm) atomicAdd(&ns[atom * AFL + colF], accm);
    }
  }
}

// ---------------- BN finalize / stats / update / pool ----------------

__global__ void k_bn1_finalize(const float* __restrict__ S, const float* __restrict__ Q,
                               const float* __restrict__ gamma, const float* __restrict__ beta,
                               const float* __restrict__ bias,
                               float* __restrict__ A1, float* __restrict__ B1) {
  int c = threadIdx.x;                       // 128
  const float inv = 1.f / (float)NROWS;
  float m = S[c] * inv;
  float v = Q[c] * inv - m * m;
  float a = gamma[c] * rsqrtf(v + EPSBN);
  A1[c] = a;
  B1[c] = a * (bias[c] - m) + beta[c];
}

__global__ void k_bn2_stats(const float* __restrict__ ns, float* __restrict__ gS,
                            float* __restrict__ gQ) {
  __shared__ float sS[AFL], sQ[AFL];
  int tid = threadIdx.x;
  if (tid < AFL) { sS[tid] = 0.f; sQ[tid] = 0.f; }
  __syncthreads();
  int c = (blockIdx.x * blockDim.x + tid) & (AFL - 1);
  float s = 0.f, q = 0.f;
  for (int i = blockIdx.x * blockDim.x + tid; i < NATOMS * AFL; i += gridDim.x * blockDim.x) {
    float v = ns[i]; s += v; q += v * v;
  }
  atomicAdd(&sS[c], s); atomicAdd(&sQ[c], q);
  __syncthreads();
  if (tid < AFL) { atomicAdd(&gS[tid], sS[tid]); atomicAdd(&gQ[tid], sQ[tid]); }
}

__global__ void k_bn2_finalize(const float* __restrict__ S, const float* __restrict__ Q,
                               const float* __restrict__ gamma, const float* __restrict__ beta,
                               float* __restrict__ A2, float* __restrict__ B2) {
  int c = threadIdx.x;                       // 64
  const float inv = 1.f / (float)NATOMS;
  float m = S[c] * inv;
  float v = Q[c] * inv - m * m;
  float a = gamma[c] * rsqrtf(v + EPSBN);
  A2[c] = a;
  B2[c] = beta[c] - m * a;
}

__global__ void k_update(float* __restrict__ x, const float* __restrict__ ns,
                         const float* __restrict__ A2, const float* __restrict__ B2) {
  int i = blockIdx.x * blockDim.x + threadIdx.x;
  if (i >= NATOMS * AFL) return;
  int c = i & (AFL - 1);
  x[i] = softplusf(x[i] + A2[c] * ns[i] + B2[c]);
}

__global__ void k_pool(const float* __restrict__ x, const int* __restrict__ cidx,
                       const float* __restrict__ Wp, const float* __restrict__ bp,
                       float* __restrict__ out) {
  __shared__ float cf[AFL];
  int c = blockIdx.x, t = threadIdx.x;       // 128 threads / crystal
  if (t < AFL) {
    float s = 0.f;
    for (int a = 0; a < APC; ++a) s += x[(size_t)cidx[c * APC + a] * AFL + t];
    cf[t] = s * (1.f / (float)APC);
  }
  __syncthreads();
  float d = bp[t];
  for (int k = 0; k < AFL; ++k) d += cf[k] * Wp[k * ODIM + t];
  out[(size_t)c * ODIM + t] = d > 0.f ? d : 0.f;
}

// ---------------- launch ----------------

extern "C" void kernel_launch(void* const* d_in, const int* in_sizes, int n_in,
                              void* d_out, int out_size, void* d_ws, size_t ws_size,
                              hipStream_t stream) {
  const float* atom_fea  = (const float*)d_in[0];
  const float* nbr_fea   = (const float*)d_in[1];
  const int*   nbr_idx   = (const int*)d_in[2];
  const int*   cry_idx   = (const int*)d_in[3];
  const float* W_embed   = (const float*)d_in[4];
  const float* b_embed   = (const float*)d_in[5];
  const float* W_full    = (const float*)d_in[6];   // [3][169][128]
  const float* b_full    = (const float*)d_in[7];   // [3][128]
  const float* bn1_gamma = (const float*)d_in[8];
  const float* bn1_beta  = (const float*)d_in[9];
  const float* bn2_gamma = (const float*)d_in[10];
  const float* bn2_beta  = (const float*)d_in[11];
  const float* W_pool    = (const float*)d_in[12];
  const float* b_pool    = (const float*)d_in[13];
  float* out = (float*)d_out;

  char* w = (char*)d_ws;
  size_t off = 0;
  float* x  = (float*)(w + off); off += (size_t)NATOMS * AFL * 4;          // 25.6 MB
  float* ns = (float*)(w + off); off += (size_t)NATOMS * AFL * 4;          // 25.6 MB
  unsigned* WeT = (unsigned*)(w + off); off += (size_t)WE_DW * 4;          // 12 KB
  unsigned* WfT = (unsigned*)(w + off); off += (size_t)WF_DW * 4;          // 48 KB
  float* bn1S = (float*)(w + off); off += 512;
  float* bn1Q = (float*)(w + off); off += 512;
  float* bn2S = (float*)(w + off); off += 512;
  float* bn2Q = (float*)(w + off); off += 512;
  float* A1v  = (float*)(w + off); off += 512;
  float* B1v  = (float*)(w + off); off += 512;
  float* A2v  = (float*)(w + off); off += 512;
  float* B2v  = (float*)(w + off); off += 512;

  // embedding
  k_cvt_wemb<<<12, 256, 0, stream>>>(W_embed, WeT);
  k_embed<<<(NATOMS + 127) / 128, 256, 0, stream>>>(atom_fea, WeT, b_embed, x);

  for (int l = 0; l < 3; ++l) {
    k_cvt_wfull<<<48, 256, 0, stream>>>(W_full + (size_t)l * INFULL * 2 * AFL, WfT);
    hipMemsetAsync(bn1S, 0, 4 * 512, stream);                    // bn1S..bn2Q contiguous
    hipMemsetAsync(ns, 0, (size_t)NATOMS * AFL * 4, stream);

    const float* bias = b_full + l * 2 * AFL;
    k_conv<false><<<NROWS / 128, 256, 0, stream>>>(x, nbr_idx, nbr_fea, WfT, bias,
                                                   A1v, B1v, bn1S, bn1Q, ns);
    k_bn1_finalize<<<1, 128, 0, stream>>>(bn1S, bn1Q, bn1_gamma + l * 2 * AFL,
                                          bn1_beta + l * 2 * AFL, bias, A1v, B1v);
    k_conv<true><<<NROWS / 128, 256, 0, stream>>>(x, nbr_idx, nbr_fea, WfT, bias,
                                                  A1v, B1v, bn1S, bn1Q, ns);
    k_bn2_stats<<<400, 256, 0, stream>>>(ns, bn2S, bn2Q);
    k_bn2_finalize<<<1, 64, 0, stream>>>(bn2S, bn2Q, bn2_gamma + l * AFL,
                                         bn2_beta + l * AFL, A2v, B2v);
    k_update<<<(NATOMS * AFL) / 256, 256, 0, stream>>>(x, ns, A2v, B2v);
  }

  k_pool<<<NCRY, 128, 0, stream>>>(x, cry_idx, W_pool, b_pool, out);
}
